// PropAttention_14791867367709
// MI455X (gfx1250) — matrix operand order repaired and checked by reading, not compiled, not run on hardware
//
#include <hip/hip_runtime.h>
#include <stdint.h>

// ---------- CDNA5 WMMA types ----------
typedef __attribute__((ext_vector_type(16))) __bf16 v16bf;
typedef __attribute__((ext_vector_type(8)))  float  v8f;

union Frag { uint4 q[2]; v16bf v; };      // 16 bf16 = 32B = one wave32 A/B fragment
union U4   { uint4 q; unsigned short s[8]; };

__device__ __forceinline__ unsigned short f2bf(float f) {
  unsigned int u = __float_as_uint(f);
  unsigned int r = (u + 0x7FFFu + ((u >> 16) & 1u)) >> 16;  // round-nearest-even
  return (unsigned short)r;
}

__device__ __forceinline__ v8f wmma_bf16(const Frag& a, const Frag& b, v8f c) {
  // D = A(16x32 bf16) * B(32x16 bf16) + C(16x16 f32)
  return __builtin_amdgcn_wmma_f32_16x16x32_bf16(false, a.v, false, b.v,
                                                 (short)0, c, false, false);
}

// ---------- TDM (Tensor Data Mover) 2D tile load, CDNA5 ISA ----------
#if defined(__AMDGCN__) && __has_builtin(__builtin_amdgcn_tensor_load_to_lds) && \
    __has_builtin(__builtin_amdgcn_s_wait_tensorcnt)
#define HAVE_TDM 1
typedef __attribute__((ext_vector_type(4))) unsigned int u32x4;
typedef __attribute__((ext_vector_type(8))) int i32x8;
typedef __attribute__((ext_vector_type(4))) int i32x4;

// Load a tile_m x tile_k (elements, 2B each) tile of a row-major matrix
// (row stride = stride_e elements) into LDS, packed row-major [tile_m][tile_k].
__device__ __forceinline__ void tdm_load_2d_bf16(const unsigned short* gsrc,
                                                 unsigned lds_byte_off,
                                                 unsigned tile_k, unsigned tile_m,
                                                 unsigned tensor_k, unsigned tensor_m,
                                                 unsigned long long stride_e) {
  unsigned long long ga = (unsigned long long)(uintptr_t)gsrc;
  u32x4 g0;
  g0[0] = 1u;                                         // count=1 (valid), user mode
  g0[1] = lds_byte_off;                               // lds_addr
  g0[2] = (unsigned)(ga & 0xFFFFFFFFu);               // global_addr[31:0]
  g0[3] = (unsigned)((ga >> 32) & 0x01FFFFFFu)        // global_addr[56:32]
        | (2u << 30);                                 // type = 2 ("image")
  i32x8 g1;
  g1[0] = (int)(1u << 16);                            // data_size=1 (2 bytes)
  g1[1] = (int)((tensor_k & 0xFFFFu) << 16);          // tensor_dim0[15:0] @ bits63:48
  g1[2] = (int)(((tensor_k >> 16) & 0xFFFFu) |        // tensor_dim0[31:16]
                ((tensor_m & 0xFFFFu) << 16));        // tensor_dim1[15:0]
  g1[3] = (int)(((tensor_m >> 16) & 0xFFFFu) |        // tensor_dim1[31:16]
                ((tile_k & 0xFFFFu) << 16));          // tile_dim0
  g1[4] = (int)(tile_m & 0xFFFFu);                    // tile_dim1 (tile_dim2 = 0)
  g1[5] = (int)(unsigned)(stride_e & 0xFFFFFFFFull);  // tensor_dim0_stride[31:0]
  g1[6] = (int)(unsigned)((stride_e >> 32) & 0xFFFFull); // [47:32]; dim1_stride=0
  g1[7] = 0;
  i32x4 z4; z4[0] = 0; z4[1] = 0; z4[2] = 0; z4[3] = 0;  // groups 2/3 unused (2D)
  i32x8 z8;
#pragma unroll
  for (int i = 0; i < 8; i++) z8[i] = 0;
  // 6-arg toolchain variant: (g0, g1, g2, g3, extra, cpol)
  __builtin_amdgcn_tensor_load_to_lds(g0, g1, z4, z4, z8, 0);
}

__device__ __forceinline__ unsigned lds_offset(const void* p) {
  // generic LDS pointer: aperture base in [63:32], byte offset in [31:0]
  return (unsigned)(uintptr_t)p;
}
#else
#define HAVE_TDM 0
#endif

// ---------- f32 -> bf16 conversion ----------
__global__ void cvt_f32_bf16(const float* __restrict__ src,
                             unsigned short* __restrict__ dst, int n) {
  int i = blockIdx.x * blockDim.x + threadIdx.x;
  int stride = gridDim.x * blockDim.x;
  for (; i < n; i += stride) dst[i] = f2bf(src[i]);
}

// ---------- 128x128x32-tiled bf16 GEMM, f32 accumulate, double-buffered ----------
// C[M,N] = A[M,K] @ B[K,N]; M,N multiples of 128, K multiple of 32.
// A tiles staged by the Tensor Data Mover (TENSORcnt), B tiles staged transposed
// by the threads. OUT_F32: write float (+bias), else bf16.
template <bool OUT_F32>
__global__ __launch_bounds__(256)
void gemm_bf16_128x128(const unsigned short* __restrict__ A,
                       const unsigned short* __restrict__ B,
                       void* __restrict__ C,
                       const float* __restrict__ bias,
                       int M, int N, int K) {
  __shared__ unsigned short As[2][128][32];   // [m][k]
  __shared__ unsigned short Bs[2][128][32];   // transposed: [n][k]

  const int m0 = blockIdx.x * 128;
  const int n0 = blockIdx.y * 128;
  const int t  = threadIdx.x;
  const int wave = t >> 5, lane = t & 31;
  const int l15 = lane & 15, hi = lane >> 4;
  const int wr = wave >> 2, wc = wave & 3;   // 2x4 wave grid

  v8f acc[4][2];
#pragma unroll
  for (int i = 0; i < 4; i++)
#pragma unroll
    for (int j = 0; j < 2; j++) acc[i][j] = (v8f)0.0f;

  const int bk = t >> 3, bn = (t & 7) * 16;       // B staging: 16 bf16/thread
#if !HAVE_TDM
  const int arow = t >> 1, acol = (t & 1) * 16;   // A staging fallback
#endif

  auto stage = [&](int k0, int q) {
#if HAVE_TDM
    if (wave == 0) {   // wave-uniform: one TDM descriptor per tile
      tdm_load_2d_bf16(A + (size_t)m0 * K + k0, lds_offset(&As[q][0][0]),
                       32u, 128u, (unsigned)K, (unsigned)M,
                       (unsigned long long)K);
    }
#else
    const uint4* pa = (const uint4*)(A + (size_t)(m0 + arow) * K + k0 + acol);
    uint4 a0 = pa[0], a1 = pa[1];
    *(uint4*)&As[q][arow][acol]     = a0;
    *(uint4*)&As[q][arow][acol + 8] = a1;
#endif
    // B tile transposed: coalesced global b128, scattered ds_store_b16
    U4 b0, b1;
    const uint4* pb = (const uint4*)(B + (size_t)(k0 + bk) * N + n0 + bn);
    b0.q = pb[0]; b1.q = pb[1];
#pragma unroll
    for (int j = 0; j < 8; j++) {
      Bs[q][bn + j][bk]     = b0.s[j];
      Bs[q][bn + 8 + j][bk] = b1.s[j];
    }
  };

  stage(0, 0);
#if HAVE_TDM
  if (wave == 0) __builtin_amdgcn_s_wait_tensorcnt(0);
#endif
  __syncthreads();

  int p = 0;
  for (int k0 = 0; k0 < K; k0 += 32) {
    const int kn = k0 + 32;
    if (kn < K) stage(kn, p ^ 1);   // prefetch next tile into the other buffer

    // fragments (CDNA5 ISA 7.12.2 layouts)
    Frag af[4], bf[2];
#pragma unroll
    for (int ms = 0; ms < 4; ms++) {
      int row = wr * 64 + ms * 16 + l15;   // A: lane<16 -> M=lane, K in {0..7,16..23}
      int kg = hi * 8;
      af[ms].q[0] = *(const uint4*)&As[p][row][kg];
      af[ms].q[1] = *(const uint4*)&As[p][row][kg + 16];
    }
#pragma unroll
    for (int ns = 0; ns < 2; ns++) {
      int n = wc * 32 + ns * 16 + l15;     // B: lane holds column n, K in {hi*8..hi*8+7, 16+hi*8..}
      int ko = hi * 8;
      bf[ns].q[0] = *(const uint4*)&Bs[p][n][ko];
      bf[ns].q[1] = *(const uint4*)&Bs[p][n][ko + 16];
    }
#pragma unroll
    for (int ms = 0; ms < 4; ms++)
#pragma unroll
      for (int ns = 0; ns < 2; ns++)
        acc[ms][ns] = wmma_bf16(af[ms], bf[ns], acc[ms][ns]);

#if HAVE_TDM
    if (wave == 0 && kn < K) __builtin_amdgcn_s_wait_tensorcnt(0);
#endif
    __syncthreads();
    p ^= 1;
  }

  // epilogue: C/D layout -> VGPR r holds (M = r + hi*8, N = l15)
#pragma unroll
  for (int ms = 0; ms < 4; ms++) {
#pragma unroll
    for (int ns = 0; ns < 2; ns++) {
      int col = n0 + wc * 32 + ns * 16 + l15;
      int rowbase = m0 + wr * 64 + ms * 16 + hi * 8;
      float bv = 0.0f;
      if (OUT_F32 && bias) bv = bias[col];
#pragma unroll
      for (int r = 0; r < 8; r++) {
        float v = acc[ms][ns][r];
        size_t off = (size_t)(rowbase + r) * N + col;
        if (OUT_F32) ((float*)C)[off] = v + bv;
        else ((unsigned short*)C)[off] = f2bf(v);
      }
    }
  }
}

// ---------- gathered per-stack attention ----------
// One workgroup (6 waves) per stack b (768). K rows + transposed V in LDS,
// Q fragments gathered from global. Output scattered into permuted (96,768,512).
#define KP 104   // padded key stride for transposed V

__global__ __launch_bounds__(192)
void attn_stacks(const unsigned short* __restrict__ Qg,    // [9216][512]
                 const unsigned short* __restrict__ KVg,   // [9216][1024] (k|v)
                 const int* __restrict__ gidx,             // [768][96]
                 unsigned short* __restrict__ Y) {         // [73728][512] bf16
  extern __shared__ unsigned short smem[];
  unsigned short* Ks = smem;            // [96][512]  key rows (row-major)
  unsigned short* Vt = Ks + 96 * 512;   // [512][KP]  V transposed: [dim][key]
  unsigned short* Ps = Vt + 512 * KP;   // [6][16][96] per-wave P scratch

  const int b  = blockIdx.x;
  const int lb = b >> 3, h = b & 7;
  const int t  = threadIdx.x;

  // ---- gather-stage K and V(transposed) into LDS ----
  {
    int row = t >> 1, which = t & 1;            // 96 rows x {K,V}
    int g = gidx[b * 96 + row];
    const uint4* src = (const uint4*)(KVg + (size_t)g * 1024 + which * 512);
    if (which == 0) {
      uint4* dst = (uint4*)&Ks[row * 512];
#pragma unroll 4
      for (int i = 0; i < 64; i++) dst[i] = src[i];
    } else {
      for (int i = 0; i < 64; i++) {
        U4 u; u.q = src[i];
#pragma unroll
        for (int j = 0; j < 8; j++) Vt[(size_t)(i * 8 + j) * KP + row] = u.s[j];
      }
    }
  }
  __syncthreads();

  const int wave = t >> 5, lane = t & 31;
  const int l15 = lane & 15, hi = lane >> 4;

  // per-lane gathered Q row for this wave's 16-query tile
  int qrow = gidx[b * 96 + wave * 16 + l15];
  const unsigned short* qptr = Qg + (size_t)qrow * 512;

  // ---- scores: 6 key tiles x 16 K-steps ----
  v8f sc[6];
#pragma unroll
  for (int i = 0; i < 6; i++) sc[i] = (v8f)0.0f;

  for (int kt = 0; kt < 16; kt++) {
    Frag af;
    int kg = kt * 32 + hi * 8;
    af.q[0] = *(const uint4*)(qptr + kg);
    af.q[1] = *(const uint4*)(qptr + kg + 16);
#pragma unroll
    for (int nt = 0; nt < 6; nt++) {
      Frag bfr;
      int ko = kt * 32 + hi * 8;
      const unsigned short* kp = &Ks[(nt * 16 + l15) * 512 + ko];
      bfr.q[0] = *(const uint4*)kp;
      bfr.q[1] = *(const uint4*)(kp + 16);
      sc[nt] = wmma_bf16(af, bfr, sc[nt]);
    }
  }

  // ---- softmax over keys (rows live in VGPR r, cols striped over 16-lane halves) ----
#pragma unroll
  for (int nt = 0; nt < 6; nt++)
#pragma unroll
    for (int r = 0; r < 8; r++) sc[nt][r] *= 0.125f;   // 64^-0.5

  unsigned short* pw = &Ps[wave * 16 * 96];
#pragma unroll
  for (int r = 0; r < 8; r++) {
    float mx = -3.0e38f;
#pragma unroll
    for (int nt = 0; nt < 6; nt++) mx = fmaxf(mx, sc[nt][r]);
#pragma unroll
    for (int m = 1; m < 16; m <<= 1) mx = fmaxf(mx, __shfl_xor(mx, m, 32));
    float e[6]; float sum = 0.0f;
#pragma unroll
    for (int nt = 0; nt < 6; nt++) { e[nt] = __expf(sc[nt][r] - mx); sum += e[nt]; }
#pragma unroll
    for (int m = 1; m < 16; m <<= 1) sum += __shfl_xor(sum, m, 32);
    float inv = 1.0f / sum;
    int prow = r + hi * 8;
#pragma unroll
    for (int nt = 0; nt < 6; nt++)
      pw[prow * 96 + nt * 16 + l15] = f2bf(e[nt] * inv);
  }

  // ---- re-layout P (C-layout -> A-layout) via wave-local LDS round trip ----
  Frag pf[3];
#pragma unroll
  for (int ks = 0; ks < 3; ks++) {
    int kg = ks * 32 + hi * 8;
    const unsigned short* pp = &pw[l15 * 96 + kg];
    pf[ks].q[0] = *(const uint4*)pp;
    pf[ks].q[1] = *(const uint4*)(pp + 16);
  }

  // ---- PV: out(16 x 512) in 32 dim-chunks, scatter to permuted layout ----
  for (int nc = 0; nc < 32; nc++) {
    v8f o = (v8f)0.0f;
#pragma unroll
    for (int ks = 0; ks < 3; ks++) {
      Frag vf;
      int ko = ks * 32 + hi * 8;
      const unsigned short* vp = &Vt[(size_t)(nc * 16 + l15) * KP + ko];
      vf.q[0] = *(const uint4*)vp;
      vf.q[1] = *(const uint4*)(vp + 16);
      o = wmma_bf16(pf[ks], vf, o);
    }
    int e = nc * 16 + l15;
    int ehi = e >> 6, col = h * 64 + (e & 63);
#pragma unroll
    for (int r = 0; r < 8; r++) {
      int i = wave * 16 + hi * 8 + r;
      size_t yrow = (size_t)lb * 768 + (size_t)i * 8 + ehi;  // permuted reshape
      Y[yrow * 512 + col] = f2bf(o[r]);
    }
  }
}

// ---------- host side ----------
extern "C" void kernel_launch(void* const* d_in, const int* in_sizes, int n_in,
                              void* d_out, int out_size, void* d_ws, size_t ws_size,
                              hipStream_t stream) {
  (void)in_sizes; (void)n_in; (void)out_size; (void)ws_size;
  const float* x    = (const float*)d_in[0];   // (96,96,512)
  const float* Wq   = (const float*)d_in[1];   // (512,512)
  const float* Wkv  = (const float*)d_in[2];   // (512,1024)
  const float* Wout = (const float*)d_in[3];   // (512,512)
  const float* bout = (const float*)d_in[4];   // (512,)
  const int*   gidx = (const int*)d_in[5];     // (768,96)

  unsigned short* ws    = (unsigned short*)d_ws;
  unsigned short* Xb    = ws;                       // 9216*512
  unsigned short* Wqb   = Xb    + 9216 * 512;       // 512*512
  unsigned short* Wkvb  = Wqb   + 512 * 512;        // 512*1024
  unsigned short* Woutb = Wkvb  + 512 * 1024;       // 512*512
  unsigned short* Qb    = Woutb + 512 * 512;        // 9216*512
  unsigned short* KVb   = Qb    + 9216 * 512;       // 9216*1024
  unsigned short* Yb    = KVb   + 9216 * 1024;      // 73728*512

  cvt_f32_bf16<<<dim3(1024), dim3(256), 0, stream>>>(x,    Xb,    9216 * 512);
  cvt_f32_bf16<<<dim3(256),  dim3(256), 0, stream>>>(Wq,   Wqb,   512 * 512);
  cvt_f32_bf16<<<dim3(512),  dim3(256), 0, stream>>>(Wkv,  Wkvb,  512 * 1024);
  cvt_f32_bf16<<<dim3(256),  dim3(256), 0, stream>>>(Wout, Woutb, 512 * 512);

  // Q = X @ Wq ; KV = X @ Wkv
  gemm_bf16_128x128<false><<<dim3(72, 4), dim3(256), 0, stream>>>(
      Xb, Wqb, Qb, nullptr, 9216, 512, 512);
  gemm_bf16_128x128<false><<<dim3(72, 8), dim3(256), 0, stream>>>(
      Xb, Wkvb, KVb, nullptr, 9216, 1024, 512);

  // per-stack attention with gather + permuted output
  size_t smem = (size_t)(96 * 512 + 512 * KP + 6 * 16 * 96) * sizeof(unsigned short);
  attn_stacks<<<dim3(768), dim3(192), smem, stream>>>(Qb, KVb, gidx, Yb);

  // final projection: d_out = Y @ Wout + bout (f32 output)
  gemm_bf16_128x128<true><<<dim3(576, 4), dim3(256), 0, stream>>>(
      Yb, Woutb, d_out, bout, 73728, 512, 512);
}
